// NeuralMapCell_63144609186312
// MI455X (gfx1250) — matrix operand order repaired and checked by reading, not compiled, not run on hardware
//
#include <hip/hip_runtime.h>
#include <cstddef>

// ---------------------------------------------------------------------------
// MI455X (gfx1250) NeuralMapCell. Matrix work via V_WMMA_F32_16X16X32_BF16
// (wave32, 8 waves / 256-thread workgroups). GEMM A-slab staging uses the
// Tensor Data Mover (TENSOR_LOAD_TO_LDS + s_wait_tensorcnt) via the 6-arg
// clang-23 builtin, else manual LDS staging. Memory-bound stages (attention
// einsums + fused new_memory copy) are coalesced streaming.
// B=1024, UNITS=256, H=W=16.
// ---------------------------------------------------------------------------

typedef __attribute__((ext_vector_type(16))) __bf16 v16bf;
typedef __attribute__((ext_vector_type(8)))  float  v8f;

#if defined(__gfx1250__) && __has_builtin(__builtin_amdgcn_tensor_load_to_lds)
#define USE_TDM 1
typedef __attribute__((ext_vector_type(4))) unsigned int u32x4;
typedef __attribute__((ext_vector_type(8))) int          i32x8;
typedef __attribute__((ext_vector_type(4))) int          i32x4;
#endif

union FragAB {
  v16bf v;
  unsigned int   w[8];
  unsigned short u[16];
};

__device__ inline unsigned short f2bf(float f) {
  unsigned int u = __float_as_uint(f);
  unsigned int r = u + 0x7FFFu + ((u >> 16) & 1u);   // round-to-nearest-even
  return (unsigned short)(r >> 16);
}

// ISA 16-bit A-matrix 16x32 layout: per-lane K offsets come in even pairs.
// lanes 0-15 hold K {0..7,16..23}; lanes 16-31 hold K {8..15,24..31}.
__device__ inline int a_koff_base(int lhalf, int t) {
  return lhalf ? ((t < 4) ? 2 * t + 8 : 2 * t + 16)
               : ((t < 4) ? 2 * t     : 2 * t + 8);
}
// B-matrix 32x16: same K order as A; lanes 0-15 K {0..7,16..23}, lanes 16-31
// K {8..15,24..31}, column N = lane%16.
__device__ inline int b_koff_base(int lhalf, int t) {
  return a_koff_base(lhalf, t);
}

// ------------------------------ converts -----------------------------------
__global__ void cvt_bf16_kernel(const float* __restrict__ s,
                                unsigned short* __restrict__ d, int n) {
  int i = blockIdx.x * 256 + threadIdx.x;
  if (i < n) d[i] = f2bf(s[i]);
}

// context [1024,256] f32 -> concat buffer [1024,512] bf16 cols 0..255
__global__ void cvt_ctx_kernel(const float* __restrict__ s,
                               unsigned short* __restrict__ d, int n) {
  int i = blockIdx.x * 256 + threadIdx.x;
  if (i < n) {
    int row = i >> 8, col = i & 255;
    d[row * 512 + col] = f2bf(s[i]);
  }
}

// ------------------------------ conv1 (WMMA) -------------------------------
// h1[b][oc=32][p=256] = SAME 3x3 conv of memory[b][256][16][16] with k1.
// Implicit GEMM: M=256 pixels, N=32, K=9*256.  One workgroup per batch.
__global__ __launch_bounds__(256) void conv1_wmma(
    const float* __restrict__ mem, const unsigned short* __restrict__ k1bf,
    unsigned short* __restrict__ h1bf) {
  __shared__ unsigned short sA[256 * 32];  // [pixel][ic_local]   16 KB
  __shared__ unsigned short sB[32 * 32];   // [oc][k_local]        2 KB
  const int b = blockIdx.x;
  const int tid = threadIdx.x;
  const int w = tid >> 5, l = tid & 31;
  const int l16 = l & 15, lhalf = l >> 4;
  const float* memb = mem + (size_t)b * 65536;

  v8f acc[2][2];
#pragma unroll
  for (int i = 0; i < 2; ++i)
#pragma unroll
    for (int j = 0; j < 2; ++j)
#pragma unroll
      for (int e = 0; e < 8; ++e) acc[i][j][e] = 0.f;

  for (int kb = 0; kb < 72; ++kb) {
    const int tap = kb >> 3;
    const int ic0 = (kb & 7) << 5;
    const int dy = tap / 3 - 1, dx = tap % 3 - 1;
    __syncthreads();
    // stage memory slab (bf16, transposed to [pixel][ic]) — coalesced loads
#pragma unroll 4
    for (int it = 0; it < 32; ++it)
      sA[tid * 32 + it] = f2bf(memb[(ic0 + it) * 256 + tid]);
    // stage weight slab transposed to [oc][k]
    for (int idx = tid; idx < 1024; idx += 256) {
      int k = idx & 31, oc = idx >> 5;
      sB[oc * 32 + k] = k1bf[(tap * 256 + ic0 + k) * 32 + oc];
    }
    __syncthreads();

    FragAB bfr[2];
#pragma unroll
    for (int nt = 0; nt < 2; ++nt) {
      const unsigned short* bp = &sB[(nt * 16 + l16) * 32];
#pragma unroll
      for (int t = 0; t < 8; ++t)
        bfr[nt].w[t] = *(const unsigned int*)(bp + b_koff_base(lhalf, t));
    }
#pragma unroll
    for (int mt = 0; mt < 2; ++mt) {
      int p = (w * 2 + mt) * 16 + l16;
      int sy = (p >> 4) + dy, sx = (p & 15) + dx;
      bool inb = (sy >= 0) & (sy < 16) & (sx >= 0) & (sx < 16);
      const unsigned short* ap = &sA[((inb ? sy : 0) * 16 + (inb ? sx : 0)) * 32];
      FragAB afr;
#pragma unroll
      for (int t = 0; t < 8; ++t)
        afr.w[t] = inb ? *(const unsigned int*)(ap + a_koff_base(lhalf, t)) : 0u;
#pragma unroll
      for (int nt = 0; nt < 2; ++nt)
        acc[mt][nt] = __builtin_amdgcn_wmma_f32_16x16x32_bf16(
            false, afr.v, false, bfr[nt].v, (short)0, acc[mt][nt], false, false);
    }
  }
  unsigned short* out = h1bf + (size_t)b * 8192;
#pragma unroll
  for (int mt = 0; mt < 2; ++mt)
#pragma unroll
    for (int nt = 0; nt < 2; ++nt)
#pragma unroll
      for (int e = 0; e < 8; ++e) {
        int p = (w * 2 + mt) * 16 + e + 8 * lhalf;
        int oc = nt * 16 + l16;
        out[oc * 256 + p] = f2bf(acc[mt][nt][e]);
      }
}

// ------------------------------ conv2 (WMMA) -------------------------------
// h2[b][oc=64][q=196] = VALID 3x3 conv of h1.  M=196(->13 tiles), N=64, K=288.
__global__ __launch_bounds__(256) void conv2_wmma(
    const unsigned short* __restrict__ h1bf,
    const unsigned short* __restrict__ k2bf, float* __restrict__ h2) {
  __shared__ unsigned short sH[256 * 32];  // [pixel][ic]   16 KB
  __shared__ unsigned short sK[64 * 288];  // [oc][k]       36.9 KB
  const int b = blockIdx.x;
  const int tid = threadIdx.x;
  const int w = tid >> 5, l = tid & 31;
  const int l16 = l & 15, lhalf = l >> 4;
  const unsigned short* h1b = h1bf + (size_t)b * 8192;

#pragma unroll 4
  for (int it = 0; it < 32; ++it) sH[tid * 32 + it] = h1b[it * 256 + tid];
  for (int idx = tid; idx < 18432; idx += 256) {
    int oc = idx / 288, k = idx % 288;
    sK[idx] = k2bf[k * 64 + oc];  // transpose to [oc][k]
  }
  __syncthreads();

  float* outb = h2 + (size_t)b * 12544;
  for (int t = w; t < 52; t += 8) {  // 13 M-tiles x 4 N-tiles
    int mt = t % 13, nt = t / 13;
    int q = mt * 16 + l16;
    int qc = (q >= 196) ? 0 : q;     // pad rows computed on row 0, masked at store
    int oy = qc / 14, ox = qc % 14;
    v8f acc;
#pragma unroll
    for (int e = 0; e < 8; ++e) acc[e] = 0.f;
#pragma unroll
    for (int kb = 0; kb < 9; ++kb) {
      int ip = (oy + kb / 3) * 16 + (ox + kb % 3);
      const unsigned short* ap = &sH[ip * 32];
      const unsigned short* bp = &sK[(nt * 16 + l16) * 288 + kb * 32];
      FragAB afr, bfr;
#pragma unroll
      for (int tt = 0; tt < 8; ++tt) {
        afr.w[tt] = *(const unsigned int*)(ap + a_koff_base(lhalf, tt));
        bfr.w[tt] = *(const unsigned int*)(bp + b_koff_base(lhalf, tt));
      }
      acc = __builtin_amdgcn_wmma_f32_16x16x32_bf16(
          false, afr.v, false, bfr.v, (short)0, acc, false, false);
    }
#pragma unroll
    for (int e = 0; e < 8; ++e) {
      int qo = mt * 16 + e + 8 * lhalf;
      if (qo < 196) outb[(nt * 16 + l16) * 196 + qo] = acc[e];
    }
  }
}

// ------------------------------ pool + flatten -----------------------------
__global__ void pool_kernel(const float* __restrict__ h2,
                            unsigned short* __restrict__ flat) {
  int idx = blockIdx.x * 256 + threadIdx.x;  // B*10816 = 11,075,584 exactly
  int b = idx / 10816, f = idx % 10816;
  int oc = f / 169, r = f % 169, py = r / 13, px = r % 13;
  const float* hb = h2 + (size_t)b * 12544 + oc * 196;
  int q = py * 14 + px;
  flat[idx] = f2bf(0.25f * (hb[q] + hb[q + 1] + hb[q + 14] + hb[q + 15]));
}

// ------------------------ generic bf16 WMMA GEMM ---------------------------
// C[row][col] = sum_k A[row][k]*B[k][col].  128x128 tile per workgroup,
// BK=32, 8 waves each owning one 16-row stripe x 8 N-tiles (A-frag reuse x8).
// A slab (128 rows x 64B, row stride lda, contiguous LDS dest) is staged by
// the Tensor Data Mover when available.
__global__ __launch_bounds__(256) void gemm_bf16_wmma(
    const unsigned short* __restrict__ A, int lda,
    const unsigned short* __restrict__ Bm, int ldb, int K,
    float* __restrict__ Cf, int ldcf, unsigned short* __restrict__ Cb,
    int ldcb) {
  __shared__ unsigned short sA[128 * 32];   // [r][k]  8 KB
  __shared__ unsigned short sBt[128 * 32];  // [n][k]  8 KB
  const int row0 = blockIdx.x * 128, col0 = blockIdx.y * 128;
  const int tid = threadIdx.x;
  const int w = tid >> 5, l = tid & 31;
  const int l16 = l & 15, lhalf = l >> 4;

  v8f acc[8];
#pragma unroll
  for (int nt = 0; nt < 8; ++nt)
#pragma unroll
    for (int e = 0; e < 8; ++e) acc[nt][e] = 0.f;

#if defined(USE_TDM)
  // Generic LDS pointer low 32 bits == LDS aperture offset (flat addressing).
  const unsigned lds_off = (unsigned)(size_t)(void*)&sA[0];
#endif

  for (int k0 = 0; k0 < K; k0 += 32) {
    __syncthreads();
#if defined(USE_TDM)
    if (w == 0) {
      // D# for a 2D tile: tile_dim0 = 32 elems (64B), tile_dim1 = 128 rows,
      // data_size = 2B, tensor_dim0_stride = lda. type=2 ("image").
      size_t ga = (size_t)(const void*)(A + (size_t)row0 * lda + k0);
      u32x4 g0;
      g0[0] = 1u;                                        // count=1, user mode
      g0[1] = lds_off;                                   // lds_addr
      g0[2] = (unsigned)(ga & 0xffffffffu);              // global_addr[31:0]
      g0[3] = (unsigned)((ga >> 32) & 0x1ffffffu) | (2u << 30);  // addr|type=2
      i32x8 g1;
      g1[0] = (int)(1u << 16);                           // data_size=2B, mask=0
      g1[1] = (int)0xffff0000u;                          // tensor_dim0 lo16
      g1[2] = (int)0xffff7fffu;                          // dim0 hi16 | dim1 lo16
      g1[3] = (int)((32u << 16) | 0x7fffu);              // tile_dim0=32|dim1 hi16
      g1[4] = 128;                                       // tile_dim1=128
      g1[5] = lda;                                       // stride0[31:0]
      g1[6] = 0;                                         // stride0 hi | stride1 lo
      g1[7] = 0;
      i32x4 gz;
      gz[0] = 0; gz[1] = 0; gz[2] = 0; gz[3] = 0;
      i32x8 gz8;
#pragma unroll
      for (int e = 0; e < 8; ++e) gz8[e] = 0;
      // clang-23 / therock-10.0 arity: (g0, g1, g2, g3, g4, cpol)
      __builtin_amdgcn_tensor_load_to_lds(g0, g1, gz, gz, gz8, 0);
    }
#else
#pragma unroll
    for (int it = 0; it < 8; ++it) {  // A slab: 128 rows x 16 dwords
      int idx = it * 256 + tid;
      int r = idx >> 4, kk = idx & 15;
      const unsigned short* gp = A + (size_t)(row0 + r) * lda + k0 + 2 * kk;
      ((unsigned int*)sA)[r * 16 + kk] = *(const unsigned int*)gp;
      if (k0 + 32 < K) __builtin_prefetch(gp + 32, 0, 1);  // global_prefetch_b8
    }
#endif
#pragma unroll
    for (int it = 0; it < 16; ++it) {  // B slab transposed -> [n][k]
      int idx = it * 256 + tid;
      int n = idx >> 5, k = idx & 31;
      sBt[n * 32 + k] = Bm[(size_t)(k0 + k) * ldb + col0 + n];
    }
#if defined(USE_TDM)
    if (w == 0) {
#if __has_builtin(__builtin_amdgcn_s_wait_tensorcnt)
      __builtin_amdgcn_s_wait_tensorcnt((short)0);
#else
      asm volatile("s_wait_tensorcnt 0x0" ::: "memory");
#endif
    }
#endif
    __syncthreads();

    FragAB afr;
    const unsigned short* ap = &sA[(w * 16 + l16) * 32];
#pragma unroll
    for (int t = 0; t < 8; ++t)
      afr.w[t] = *(const unsigned int*)(ap + a_koff_base(lhalf, t));
#pragma unroll
    for (int nt = 0; nt < 8; ++nt) {
      FragAB bfr;
      const unsigned short* bp = &sBt[(nt * 16 + l16) * 32];
#pragma unroll
      for (int t = 0; t < 8; ++t)
        bfr.w[t] = *(const unsigned int*)(bp + b_koff_base(lhalf, t));
      acc[nt] = __builtin_amdgcn_wmma_f32_16x16x32_bf16(
          false, afr.v, false, bfr.v, (short)0, acc[nt], false, false);
    }
  }
#pragma unroll
  for (int nt = 0; nt < 8; ++nt)
#pragma unroll
    for (int e = 0; e < 8; ++e) {
      int r = row0 + w * 16 + e + 8 * lhalf;
      int c = col0 + nt * 16 + l16;
      float val = acc[nt][e];
      if (Cf) Cf[(size_t)r * ldcf + c] = val;
      if (Cb) Cb[(size_t)r * ldcb + c] = f2bf(val);
    }
}

// ------------------------------ attention ----------------------------------
// One workgroup per batch: at/c_t einsums, importance scalars, mem_t gather,
// bf16 delta-GEMM input. The memory -> new_memory copy is fused into the
// at-pass (every element of memory[b] is read exactly once there, coalesced),
// saving a full 256 MB read pass.
__global__ __launch_bounds__(256) void attn_kernel(
    const float* __restrict__ mem, const float* __restrict__ qt,
    const float* __restrict__ st, const int* __restrict__ pos,
    const float* __restrict__ rt, float* __restrict__ ct_out,
    float* __restrict__ newmem, float* __restrict__ memt,
    unsigned short* __restrict__ adel, float* __restrict__ scale) {
  __shared__ float shq[256];
  __shared__ float red[256];
  const int b = blockIdx.x, t = threadIdx.x;
  const float* mb = mem + (size_t)b * 65536;
  float* ob = newmem + (size_t)b * 65536;

  shq[t] = qt[b * 256 + t];
  __syncthreads();
  float a = 0.f;  // thread t <-> pixel t
  for (int c = 0; c < 256; ++c) {
    float v = mb[c * 256 + t];
    ob[c * 256 + t] = v;          // fused new_memory copy (coalesced)
    a += shq[c] * v;
  }
  red[t] = a;
  __syncthreads();
  for (int s = 128; s > 0; s >>= 1) {
    if (t < s) red[t] += red[t + s];
    __syncthreads();
  }
  float total = red[0];
  __syncthreads();
  shq[t] = a / total;  // normalized attention
  __syncthreads();

  float c2 = 0.f;  // thread t <-> channel t
  for (int hw = 0; hw < 256; ++hw) c2 += shq[hw] * mb[t * 256 + hw];
  ct_out[b * 256 + t] = c2;

  float s = st[b * 256 + t];
  float r = rt[b * 256 + t];
  red[t] = s * r;
  __syncthreads();
  for (int ss = 128; ss > 0; ss >>= 1) {
    if (t < ss) red[t] += red[t + ss];
    __syncthreads();
  }
  float g = red[0];
  __syncthreads();
  red[t] = s * c2;
  __syncthreads();
  for (int ss = 128; ss > 0; ss >>= 1) {
    if (t < ss) red[t] += red[t + ss];
    __syncthreads();
  }
  float lo = red[0];

  int x = pos[2 * b], y = pos[2 * b + 1];
  float m = mb[t * 256 + y * 16 + x];
  memt[b * 256 + t] = m;
  adel[b * 256 + t] = f2bf(m - s);
  if (t == 0) scale[b] = lo / (lo + g);
}

__global__ void scatter_kernel(const float* __restrict__ memt,
                               const float* __restrict__ delta,
                               const float* __restrict__ scale,
                               const int* __restrict__ pos,
                               float* __restrict__ newmem) {
  int b = blockIdx.x, c = threadIdx.x;
  int x = pos[2 * b], y = pos[2 * b + 1];
  newmem[(size_t)b * 65536 + c * 256 + y * 16 + x] =
      memt[b * 256 + c] + scale[b] * delta[b * 256 + c];
}

// ------------------------------ launcher -----------------------------------
extern "C" void kernel_launch(void* const* d_in, const int* in_sizes, int n_in,
                              void* d_out, int out_size, void* d_ws,
                              size_t ws_size, hipStream_t stream) {
  (void)in_sizes; (void)n_in; (void)out_size; (void)ws_size;
  const float* context = (const float*)d_in[0];
  const float* memory  = (const float*)d_in[1];
  const int*   pos     = (const int*)d_in[2];
  const float* k1      = (const float*)d_in[3];
  const float* k2      = (const float*)d_in[4];
  const float* d1      = (const float*)d_in[5];
  const float* d2      = (const float*)d_in[6];
  const float* ck      = (const float*)d_in[7];
  const float* rk      = (const float*)d_in[8];

  float* out    = (float*)d_out;
  float* ct_out = out;             // [1024,256]
  float* rt_out = out + 262144;    // [1024,256]
  float* nm_out = out + 524288;    // [1024,256,16,16]

  char* ws = (char*)d_ws;
  size_t off = 0;
  auto walloc = [&](size_t bytes) -> char* {
    char* p = ws + off;
    off = (off + bytes + 255) & ~(size_t)255;
    return p;
  };
  unsigned short* k1bf   = (unsigned short*)walloc(73728u * 2);
  unsigned short* k2bf   = (unsigned short*)walloc(18432u * 2);
  unsigned short* d1bf   = (unsigned short*)walloc(1384448u * 2);
  unsigned short* d2bf   = (unsigned short*)walloc(32768u * 2);
  unsigned short* ckbf   = (unsigned short*)walloc(131072u * 2);
  unsigned short* wkbf   = (unsigned short*)walloc(65536u * 2);
  unsigned short* wubf   = (unsigned short*)walloc(65536u * 2);
  unsigned short* concat = (unsigned short*)walloc(1024u * 512 * 2);
  unsigned short* h1bf   = (unsigned short*)walloc(1024u * 8192 * 2);
  float*          h2f    = (float*)walloc((size_t)1024 * 12544 * 4);
  unsigned short* flatbf = (unsigned short*)walloc((size_t)1024 * 10816 * 2);
  unsigned short* t1bf   = (unsigned short*)walloc(1024u * 128 * 2);
  float*          qtf    = (float*)walloc(262144u * 4);
  float*          stf    = (float*)walloc(262144u * 4);
  unsigned short* adel   = (unsigned short*)walloc(262144u * 2);
  float*          memt   = (float*)walloc(262144u * 4);
  float*          deltaf = (float*)walloc(262144u * 4);
  float*          scale  = (float*)walloc(1024u * 4);

  auto cvt = [&](const float* s, unsigned short* d, int n) {
    cvt_bf16_kernel<<<(n + 255) / 256, 256, 0, stream>>>(s, d, n);
  };
  // 1) weight / context conversion to bf16
  cvt(k1, k1bf, 73728);
  cvt(k2, k2bf, 18432);
  cvt(d1, d1bf, 1384448);
  cvt(d2, d2bf, 32768);
  cvt(ck, ckbf, 131072);
  cvt(rk, wkbf, 65536);            // write_kernel  = recurr_kernel[:256]
  cvt(rk + 65536, wubf, 65536);    // write_update  = recurr_kernel[256:]
  cvt_ctx_kernel<<<(262144 + 255) / 256, 256, 0, stream>>>(context, concat,
                                                           262144);
  // 2) conv1 / conv2 / pool
  conv1_wmma<<<1024, 256, 0, stream>>>(memory, k1bf, h1bf);
  conv2_wmma<<<1024, 256, 0, stream>>>(h1bf, k2bf, h2f);
  pool_kernel<<<(1024 * 10816) / 256, 256, 0, stream>>>(h2f, flatbf);
  // 3) dense chain: flat @ d1 -> t1 (bf16), t1 @ d2 -> r_t (f32 + bf16 concat)
  gemm_bf16_wmma<<<dim3(8, 1), 256, 0, stream>>>(flatbf, 10816, d1bf, 128,
                                                 10816, nullptr, 0, t1bf, 128);
  gemm_bf16_wmma<<<dim3(8, 2), 256, 0, stream>>>(t1bf, 128, d2bf, 256, 128,
                                                 rt_out, 256, concat + 256,
                                                 512);
  // 4) q_t = concat @ context_kernel ; s_t = context @ write_kernel
  gemm_bf16_wmma<<<dim3(8, 2), 256, 0, stream>>>(concat, 512, ckbf, 256, 512,
                                                 qtf, 256, nullptr, 0);
  gemm_bf16_wmma<<<dim3(8, 2), 256, 0, stream>>>(concat, 512, wkbf, 256, 256,
                                                 stf, 256, nullptr, 0);
  // 5) attention + importance + gather + fused memory copy
  attn_kernel<<<1024, 256, 0, stream>>>(memory, qtf, stf, pos, rt_out, ct_out,
                                        nm_out, memt, adel, scale);
  // 6) delta = (mem_t - s_t) @ write_update ; scatter the position update
  gemm_bf16_wmma<<<dim3(8, 2), 256, 0, stream>>>(adel, 256, wubf, 256, 256,
                                                 deltaf, 256, nullptr, 0);
  scatter_kernel<<<1024, 256, 0, stream>>>(memt, deltaf, scale, pos, nm_out);
}
